// MLPLinkPredictor_10024453669139
// MI455X (gfx1250) — matrix operand order repaired and checked by reading, not compiled, not run on hardware
//
#include <hip/hip_runtime.h>
#include <hip/hip_bf16.h>
#include <stdint.h>

// MI455X / gfx1250, wave32.
// score[e] = W2 . relu(W1a*h[src] + W1b*h[dst] + b1) + b2
// Restructured: per-node precompute AB[n] = [W1a*h[n]+b1 | W1b*h[n]] (bf16, 51MB,
// L2-resident), then edge phase is gather+add+relu+dot out of L2.

typedef __attribute__((ext_vector_type(16))) __bf16 v16bf;
typedef __attribute__((ext_vector_type(8)))  float  v8f;

#define N_FEAT 256
#define TWO_F  512
#define LDSK   264   // 256 + 8 pad (bf16 units) -> conflict-free ds_read_b128

static __device__ __forceinline__ unsigned short f32_to_bf16(float f) {
    union { float f; uint32_t u; } c; c.f = f;
    uint32_t u = c.u;
    uint32_t r = (u + 0x7FFFu + ((u >> 16) & 1u)) >> 16;  // round-to-nearest-even
    return (unsigned short)r;
}
static __device__ __forceinline__ float bf16lo_to_f32(uint32_t u) {
    union { uint32_t u; float f; } c; c.u = u << 16; return c.f;
}
static __device__ __forceinline__ float bf16hi_to_f32(uint32_t u) {
    union { uint32_t u; float f; } c; c.u = u & 0xFFFF0000u; return c.f;
}

// ---------------------------------------------------------------------------
// Kernel 1: Wt[n][k] bf16, n in [0,512), k in [0,256).
//   n < 256 : Wt[n][k] = W1_w[n][k]        (W1a^T columns)
//   n >= 256: Wt[n][k] = W1_w[n-256][256+k] (W1b^T columns)
// ---------------------------------------------------------------------------
__global__ void wt_prep_kernel(const float* __restrict__ W1_w,
                               unsigned short* __restrict__ Wt) {
    const int n = blockIdx.x;        // 0..511
    const int k = threadIdx.x;       // 0..255
    const int half = (n >> 8) << 8;  // 0 or 256
    float w = W1_w[(size_t)(n & 255) * TWO_F + half + k];
    Wt[(size_t)n * N_FEAT + k] = f32_to_bf16(w);
}

// ---------------------------------------------------------------------------
// Kernel 2: AB[n][j] = (h[n] . Wt[j]) + (j<256 ? b1[j] : 0), stored bf16.
// Block: 256 threads (8 waves), 32 nodes. Wave w covers columns [64w, 64w+64).
// Per wave: 2 M-tiles x 4 N-tiles of v_wmma_f32_16x16x32_bf16, K-loop of 8.
// ---------------------------------------------------------------------------
__global__ void __launch_bounds__(256) node_transform_kernel(
    const float* __restrict__ h, const float* __restrict__ W1_b,
    const unsigned short* __restrict__ Wt, unsigned short* __restrict__ AB,
    int n_nodes)
{
    __shared__ __align__(16) unsigned short lds[32 * LDSK];

    const int t     = threadIdx.x;
    const int lane  = t & 31;
    const int wave  = t >> 5;
    const int node0 = blockIdx.x * 32;

    // ---- stage 32x256 tile of h into LDS as bf16 (thread: 1 row x 32 cols) ----
    {
        const int row   = t >> 3;
        const int cbase = (t & 7) * 32;
        int node = node0 + row; if (node >= n_nodes) node = n_nodes - 1;
        const float* hp = h + (size_t)node * N_FEAT + cbase;
        unsigned int pk[16];
        #pragma unroll
        for (int c = 0; c < 32; c += 2) {
            pk[c >> 1] = (unsigned int)f32_to_bf16(hp[c]) |
                         ((unsigned int)f32_to_bf16(hp[c + 1]) << 16);
        }
        uint4* dstp = (uint4*)&lds[row * LDSK + cbase];
        #pragma unroll
        for (int i = 0; i < 4; ++i)
            dstp[i] = make_uint4(pk[4*i], pk[4*i+1], pk[4*i+2], pk[4*i+3]);
    }
    __syncthreads();

    union Frag { v16bf v; uint4 q[2]; };
    Frag afrag[2], bfrag[4];
    v8f acc[2][4] = {};

    const int col0 = wave * 64;
    const int sub  = lane & 15;          // A: row in tile / B: col in tile
    const int kb8  = (lane >> 4) * 8;    // unused (A fragments also use kb16)
    const int kb16 = (lane >> 4) * 16;   // B-fragment K sub-base

    for (int k0 = 0; k0 < N_FEAT; k0 += 32) {
        // A fragments (16-bit A 16x32 layout): lane holds K kb16..kb16+15 (matches B)
        #pragma unroll
        for (int mt = 0; mt < 2; ++mt) {
            const unsigned short* ap = &lds[(mt * 16 + sub) * LDSK + k0 + kb16];
            afrag[mt].q[0] = *(const uint4*)(ap);        // K = kb16 .. kb16+7
            afrag[mt].q[1] = *(const uint4*)(ap + 8);    // K = kb16+8 .. kb16+15
        }
        // B fragments (32x16): lane holds column col0+nt*16+sub, K = kb16..kb16+15
        #pragma unroll
        for (int nt = 0; nt < 4; ++nt) {
            const unsigned short* wp =
                Wt + (size_t)(col0 + nt * 16 + sub) * N_FEAT + k0 + kb16;
            bfrag[nt].q[0] = *(const uint4*)(wp);
            bfrag[nt].q[1] = *(const uint4*)(wp + 8);
        }
        #pragma unroll
        for (int mt = 0; mt < 2; ++mt)
            #pragma unroll
            for (int nt = 0; nt < 4; ++nt)
                acc[mt][nt] = __builtin_amdgcn_wmma_f32_16x16x32_bf16(
                    false, afrag[mt].v, false, bfrag[nt].v,
                    (short)0, acc[mt][nt], false, false);
    }

    // ---- store: D layout -> VGPR r, lane l holds M = (l>>4)*8 + r, N = l&15 ----
    const int mtop = (lane >> 4) * 8;
    #pragma unroll
    for (int nt = 0; nt < 4; ++nt) {
        const int j = col0 + nt * 16 + sub;
        const float bias = (j < N_FEAT) ? W1_b[j] : 0.0f;
        #pragma unroll
        for (int mt = 0; mt < 2; ++mt) {
            #pragma unroll
            for (int r = 0; r < 8; ++r) {
                const int node = node0 + mt * 16 + mtop + r;
                if (node < n_nodes)
                    AB[(size_t)node * TWO_F + j] = f32_to_bf16(acc[mt][nt][r] + bias);
            }
        }
    }
}

// ---------------------------------------------------------------------------
// Kernel 3: one wave32 per edge. Each lane handles 8 feature dims (b128 bf16
// loads out of L2-resident AB), relu + dot with W2, wave reduction.
// ---------------------------------------------------------------------------
__global__ void __launch_bounds__(256) edge_score_kernel(
    const unsigned short* __restrict__ AB,
    const int* __restrict__ src, const int* __restrict__ dst,
    const float* __restrict__ W2_w, const float* __restrict__ W2_b,
    float* __restrict__ out, int n_edges)
{
    const int lane = threadIdx.x & 31;
    const int e = blockIdx.x * 8 + (threadIdx.x >> 5);
    if (e >= n_edges) return;

    const int s = src[e];
    const int d = dst[e];

    const uint4 av = *(const uint4*)(AB + (size_t)s * TWO_F + lane * 8);
    const uint4 bv = *(const uint4*)(AB + (size_t)d * TWO_F + N_FEAT + lane * 8);
    const float4 w0 = *(const float4*)(W2_w + lane * 8);
    const float4 w1 = *(const float4*)(W2_w + lane * 8 + 4);

    const unsigned int aa[4] = {av.x, av.y, av.z, av.w};
    const unsigned int bb[4] = {bv.x, bv.y, bv.z, bv.w};
    const float wv[8] = {w0.x, w0.y, w0.z, w0.w, w1.x, w1.y, w1.z, w1.w};

    float acc = 0.0f;
    #pragma unroll
    for (int i = 0; i < 4; ++i) {
        float hlo = fmaxf(bf16lo_to_f32(aa[i]) + bf16lo_to_f32(bb[i]), 0.0f);
        float hhi = fmaxf(bf16hi_to_f32(aa[i]) + bf16hi_to_f32(bb[i]), 0.0f);
        acc = fmaf(hlo, wv[2 * i],     acc);
        acc = fmaf(hhi, wv[2 * i + 1], acc);
    }
    #pragma unroll
    for (int off = 16; off > 0; off >>= 1)
        acc += __shfl_xor(acc, off, 32);

    if (lane == 0) out[e] = acc + W2_b[0];
}

// ---------------------------------------------------------------------------
extern "C" void kernel_launch(void* const* d_in, const int* in_sizes, int n_in,
                              void* d_out, int out_size, void* d_ws, size_t ws_size,
                              hipStream_t stream) {
    const float* h    = (const float*)d_in[0];
    const int*   src  = (const int*)d_in[1];
    const int*   dst  = (const int*)d_in[2];
    const float* W1_w = (const float*)d_in[3];
    const float* W1_b = (const float*)d_in[4];
    const float* W2_w = (const float*)d_in[5];
    const float* W2_b = (const float*)d_in[6];
    float* out = (float*)d_out;

    const int n_nodes = in_sizes[0] / N_FEAT;   // 50000
    const int n_edges = in_sizes[1];            // 800000

    // workspace: Wt (512*256 bf16 = 256KB) then AB (n_nodes*512 bf16 ~ 51MB)
    unsigned short* Wt = (unsigned short*)d_ws;
    unsigned short* AB = Wt + (size_t)TWO_F * N_FEAT;

    wt_prep_kernel<<<TWO_F, N_FEAT, 0, stream>>>(W1_w, Wt);

    const int nblocks = (n_nodes + 31) / 32;
    node_transform_kernel<<<nblocks, 256, 0, stream>>>(h, W1_b, Wt, AB, n_nodes);

    const int eblocks = (n_edges + 7) / 8;
    edge_score_kernel<<<eblocks, 256, 0, stream>>>(AB, src, dst, W2_w, W2_b,
                                                   out, n_edges);
}
